// MoELayer_4045859193681
// MI455X (gfx1250) — matrix operand order repaired and checked by reading, not compiled, not run on hardware
//
#include <hip/hip_runtime.h>
#include <hip/hip_bf16.h>

// ---------------------------------------------------------------------------
// MoE SwiGLU MLP for gfx1250 (MI455X): gather-per-expert + fused bf16 WMMA.
//   out[t] = ( silu(x[t]@Wg_e) * (x[t]@Wu_e) ) @ Wd_e,  e = routing_mask[t]
// M_TILE=32 with 2Mx2N (stage1) / 2Mx8N (stage2) register blocking so each
// A/B fragment feeds 2-8 v_wmma ops -> lowers LDS + L2 bytes per WMMA.
// ---------------------------------------------------------------------------

#define D_MODEL   1024
#define D_FF      4096
#define NTOK      8192          // B * S = 2 * 4096
#define M_TILE    32            // tokens per workgroup strip (2 WMMA M-tiles)
#define FF_CHUNK  512           // D_FF processed per LDS stage
#define XA_STRIDE (D_MODEL + 8) // bf16 elems, padded to spread LDS banks
#define HC_STRIDE (FF_CHUNK + 8)

// dynamic-LDS partition (bytes)
#define LDS_XA_OFF  0
#define LDS_XA_SZ   (M_TILE * XA_STRIDE * 2)            // 66048
#define LDS_HC_OFF  (LDS_XA_OFF + LDS_XA_SZ)
#define LDS_HC_SZ   (M_TILE * HC_STRIDE * 2)            // 33280
#define LDS_TOK_OFF (LDS_HC_OFF + LDS_HC_SZ)
#define LDS_BYTES   (LDS_TOK_OFF + M_TILE * 4)          // ~99.5 KB (<320KB/WGP)

#define WB_ELEMS  (4u * 1024u * 1024u)            // elems per weight matrix
#define WS_IDX_OFF ((size_t)6 * WB_ELEMS * 2)     // 48 MiB of bf16 weights
#define WS_CNT_OFF (WS_IDX_OFF + (size_t)2 * NTOK * 4)

typedef __attribute__((ext_vector_type(16))) __bf16 v16bf;
typedef __attribute__((ext_vector_type(8)))  float  v8f;

union FragB16 { v16bf v; uint4 q[2]; };

__device__ __forceinline__ unsigned short f32_to_bf16(float f) {
    unsigned u = __float_as_uint(f);
    unsigned r = u + 0x7FFFu + ((u >> 16) & 1u);   // round-to-nearest-even
    return (unsigned short)(r >> 16);
}

__device__ __forceinline__ v8f wmma_bf16(const FragB16& a, const FragB16& b, v8f c) {
    return __builtin_amdgcn_wmma_f32_16x16x32_bf16(
        false, a.v, false, b.v, (short)0, c, false, false);
}

// ---------------------------------------------------------------------------
// Kernel 1: transpose-convert all 6 fp32 weight matrices to bf16 [N][K].
// B-operand loads runs of 8 consecutive K per lane (at +0 and +16) -> K-contiguous layout.
// ---------------------------------------------------------------------------
__global__ void convert_weights(const float* __restrict__ w0, const float* __restrict__ w1,
                                const float* __restrict__ w2, const float* __restrict__ w3,
                                const float* __restrict__ w4, const float* __restrict__ w5,
                                unsigned short* __restrict__ wbf) {
    __shared__ float tile[32][33];
    const int m = blockIdx.y;                       // 0..5: wg0,wu0,wd0,wg1,wu1,wd1
    const float* src = (m == 0) ? w0 : (m == 1) ? w1 : (m == 2) ? w2
                     : (m == 3) ? w3 : (m == 4) ? w4 : w5;
    const int is_down = ((m % 3) == 2);
    const int K = is_down ? D_FF : D_MODEL;         // contraction dim of the GEMM
    const int N = is_down ? D_MODEL : D_FF;         // output-column dim
    unsigned short* dst = wbf + (size_t)m * WB_ELEMS;

    const int tilesN = N >> 5;
    const int tn = blockIdx.x % tilesN, tk = blockIdx.x / tilesN;
    const int n0 = tn << 5, k0 = tk << 5;
    const int tx = threadIdx.x & 31, ty = threadIdx.x >> 5;  // 32x8

    #pragma unroll
    for (int i = 0; i < 32; i += 8)
        tile[ty + i][tx] = src[(size_t)(k0 + ty + i) * N + n0 + tx];
    __syncthreads();
    #pragma unroll
    for (int i = 0; i < 32; i += 8)
        dst[(size_t)(n0 + ty + i) * K + k0 + tx] = f32_to_bf16(tile[tx][ty + i]);
}

// ---------------------------------------------------------------------------
// Kernel 2 + 3: per-expert token compaction.
// ---------------------------------------------------------------------------
__global__ void init_counts(int* __restrict__ cnt) {
    if (threadIdx.x < 2) cnt[threadIdx.x] = 0;
}

__global__ void route_tokens(const int* __restrict__ mask, int* __restrict__ idx,
                             int* __restrict__ cnt) {
    const int t = blockIdx.x * blockDim.x + threadIdx.x;
    if (t < NTOK) {
        const int r = mask[t] & 1;
        const int p = atomicAdd(&cnt[r], 1);
        idx[r * NTOK + p] = t;
    }
}

// ---------------------------------------------------------------------------
// Kernel 4: fused SwiGLU MLP. One block = 32 tokens of one expert, 8 waves.
// ---------------------------------------------------------------------------
__global__ __launch_bounds__(256, 1)
void moe_swiglu_wmma(const float* __restrict__ x,
                     const unsigned short* __restrict__ wbf,
                     const int* __restrict__ idx, const int* __restrict__ cnt,
                     float* __restrict__ out) {
    const int e = blockIdx.y;
    const int n_tok = cnt[e];
    const int row0 = blockIdx.x * M_TILE;
    if (row0 >= n_tok) return;

    const unsigned short* wg = wbf + (size_t)(e * 3 + 0) * WB_ELEMS; // [D_FF][D_MODEL]
    const unsigned short* wu = wbf + (size_t)(e * 3 + 1) * WB_ELEMS; // [D_FF][D_MODEL]
    const unsigned short* wd = wbf + (size_t)(e * 3 + 2) * WB_ELEMS; // [D_MODEL][D_FF]

    extern __shared__ char smem[];
    unsigned short* Xa = (unsigned short*)(smem + LDS_XA_OFF); // 32 x 1032 bf16
    unsigned short* Hc = (unsigned short*)(smem + LDS_HC_OFF); // 32 x 520  bf16
    int*            stok = (int*)(smem + LDS_TOK_OFF);

    const int tid  = threadIdx.x;
    const int lane = tid & 31;
    const int w    = tid >> 5;

    if (tid < M_TILE) {
        const int rr = row0 + tid;
        stok[tid] = (rr < n_tok) ? idx[e * NTOK + rr] : -1;
    }
    __syncthreads();

    // ---- gather X rows, fp32 -> bf16, into LDS (8 threads x 128 cols per row) ----
    {
        const int r = tid >> 3;                 // 32 rows
        const int cbase = (tid & 7) * 128;      // 128 cols per thread
        const int t = stok[r];
        if (t >= 0) {
            const float4* src = (const float4*)(x + (size_t)t * D_MODEL + cbase);
            #pragma unroll 4
            for (int c = 0; c < 32; ++c) {
                const float4 f = src[c];
                uint2 p;
                p.x = (unsigned)f32_to_bf16(f.x) | ((unsigned)f32_to_bf16(f.y) << 16);
                p.y = (unsigned)f32_to_bf16(f.z) | ((unsigned)f32_to_bf16(f.w) << 16);
                *(uint2*)&Xa[r * XA_STRIDE + cbase + c * 4] = p;
            }
        } else {
            #pragma unroll 4
            for (int c = 0; c < 32; ++c)
                *(uint2*)&Xa[r * XA_STRIDE + cbase + c * 4] = make_uint2(0u, 0u);
        }
    }
    __syncthreads();

    // per-lane fragment geometry (wave32 WMMA layouts)
    const int m_row  = lane & 15;
    const int ksel_a = (lane < 16) ? 0 : 8;    // 16-bit A: lanes 0-15 K0..7/16..23
    const int ksel_b = (lane < 16) ? 0 : 8;    // 16-bit B: same K order as A (K0..7/16..23)
    const int ncol   = lane & 15;

    const unsigned short* ar0 = &Xa[m_row * XA_STRIDE + ksel_a];
    const unsigned short* ar1 = ar0 + 16 * XA_STRIDE;

    v8f oacc[2][8];                            // 32 rows x 128 cols per wave (f32)
    #pragma unroll
    for (int mi = 0; mi < 2; ++mi)
        #pragma unroll
        for (int j = 0; j < 8; ++j)
            #pragma unroll
            for (int i = 0; i < 8; ++i) oacc[mi][j][i] = 0.0f;

    for (int c = 0; c < D_FF / FF_CHUNK; ++c) {
        const int F0 = c * FF_CHUNK;

        // ---- stage 1: Hc = silu(X@Wg)*(X@Wu); wave owns 4 N-tiles, 2 at a time ----
        for (int nsub = 0; nsub < 2; ++nsub) {
            const int col0 = F0 + (w * 4 + nsub * 2) * 16;   // two adjacent N tiles
            v8f ag[2][2], au[2][2];
            #pragma unroll
            for (int mi = 0; mi < 2; ++mi)
                #pragma unroll
                for (int ni = 0; ni < 2; ++ni)
                    #pragma unroll
                    for (int i = 0; i < 8; ++i) { ag[mi][ni][i] = 0.0f; au[mi][ni][i] = 0.0f; }

            const unsigned short* bg0 = wg + (size_t)(col0 +  0 + ncol) * D_MODEL + ksel_b;
            const unsigned short* bg1 = wg + (size_t)(col0 + 16 + ncol) * D_MODEL + ksel_b;
            const unsigned short* bu0 = wu + (size_t)(col0 +  0 + ncol) * D_MODEL + ksel_b;
            const unsigned short* bu1 = wu + (size_t)(col0 + 16 + ncol) * D_MODEL + ksel_b;

            #pragma unroll 2
            for (int k = 0; k < D_MODEL; k += 32) {
                FragB16 a0, a1, g0, g1, u0, u1;
                a0.q[0] = *(const uint4*)(ar0 + k);  a0.q[1] = *(const uint4*)(ar0 + k + 16);
                a1.q[0] = *(const uint4*)(ar1 + k);  a1.q[1] = *(const uint4*)(ar1 + k + 16);
                g0.q[0] = *(const uint4*)(bg0 + k);  g0.q[1] = *(const uint4*)(bg0 + k + 16);
                g1.q[0] = *(const uint4*)(bg1 + k);  g1.q[1] = *(const uint4*)(bg1 + k + 16);
                u0.q[0] = *(const uint4*)(bu0 + k);  u0.q[1] = *(const uint4*)(bu0 + k + 16);
                u1.q[0] = *(const uint4*)(bu1 + k);  u1.q[1] = *(const uint4*)(bu1 + k + 16);
                // 8 WMMAs from 6 fragments: A reused 4x, B reused 2x
                ag[0][0] = wmma_bf16(a0, g0, ag[0][0]);
                ag[0][1] = wmma_bf16(a0, g1, ag[0][1]);
                ag[1][0] = wmma_bf16(a1, g0, ag[1][0]);
                ag[1][1] = wmma_bf16(a1, g1, ag[1][1]);
                au[0][0] = wmma_bf16(a0, u0, au[0][0]);
                au[0][1] = wmma_bf16(a0, u1, au[0][1]);
                au[1][0] = wmma_bf16(a1, u0, au[1][0]);
                au[1][1] = wmma_bf16(a1, u1, au[1][1]);
            }
            // silu(g) * u -> bf16 into Hc
            #pragma unroll
            for (int mi = 0; mi < 2; ++mi)
                #pragma unroll
                for (int ni = 0; ni < 2; ++ni)
                    #pragma unroll
                    for (int i = 0; i < 8; ++i) {
                        const float gg = ag[mi][ni][i];
                        const float s  = gg / (1.0f + __expf(-gg));
                        const float h  = s * au[mi][ni][i];
                        const int mr = mi * 16 + ((lane < 16) ? i : (i + 8));
                        Hc[mr * HC_STRIDE + (col0 - F0) + ni * 16 + ncol] = f32_to_bf16(h);
                    }
        }
        __syncthreads();

        // ---- stage 2: out += Hc @ Wd[F0:F0+512, :]; wave owns 8 N-tiles x 2 M ----
        const unsigned short* hr0 = &Hc[m_row * HC_STRIDE + ksel_a];
        const unsigned short* hr1 = hr0 + 16 * HC_STRIDE;
        for (int k = 0; k < FF_CHUNK; k += 32) {
            FragB16 a0, a1;
            a0.q[0] = *(const uint4*)(hr0 + k);  a0.q[1] = *(const uint4*)(hr0 + k + 16);
            a1.q[0] = *(const uint4*)(hr1 + k);  a1.q[1] = *(const uint4*)(hr1 + k + 16);
            #pragma unroll
            for (int j = 0; j < 8; ++j) {
                const unsigned short* bd =
                    wd + (size_t)(w * 128 + j * 16 + ncol) * D_FF + F0 + k + ksel_b;
                FragB16 b;
                b.q[0] = *(const uint4*)(bd);
                b.q[1] = *(const uint4*)(bd + 16);
                oacc[0][j] = wmma_bf16(a0, b, oacc[0][j]);   // B reused 2x, A reused 8x
                oacc[1][j] = wmma_bf16(a1, b, oacc[1][j]);
            }
        }
        __syncthreads();
    }

    // ---- epilogue: scatter f32 results to routed token rows ----
    #pragma unroll
    for (int j = 0; j < 8; ++j) {
        const int oc0 = w * 128 + j * 16;
        #pragma unroll
        for (int mi = 0; mi < 2; ++mi)
            #pragma unroll
            for (int i = 0; i < 8; ++i) {
                const int mr = mi * 16 + ((lane < 16) ? i : (i + 8));
                const int t = stok[mr];
                if (t >= 0) out[(size_t)t * D_MODEL + oc0 + ncol] = oacc[mi][j][i];
            }
    }
}

// ---------------------------------------------------------------------------
extern "C" void kernel_launch(void* const* d_in, const int* in_sizes, int n_in,
                              void* d_out, int out_size, void* d_ws, size_t ws_size,
                              hipStream_t stream) {
    (void)in_sizes; (void)n_in; (void)out_size; (void)ws_size;
    const float* x    = (const float*)d_in[0];
    const int*   mask = (const int*)d_in[1];
    const float* wg0  = (const float*)d_in[2];
    const float* wu0  = (const float*)d_in[3];
    const float* wd0  = (const float*)d_in[4];
    const float* wg1  = (const float*)d_in[5];
    const float* wu1  = (const float*)d_in[6];
    const float* wd1  = (const float*)d_in[7];
    float* out = (float*)d_out;

    char* ws = (char*)d_ws;
    unsigned short* wbf = (unsigned short*)ws;        // 48 MiB bf16 weights
    int* idxb = (int*)(ws + WS_IDX_OFF);              // int[2][8192]
    int* cntb = (int*)(ws + WS_CNT_OFF);              // int[2]

    convert_weights<<<dim3(4096, 6), 256, 0, stream>>>(wg0, wu0, wd0, wg1, wu1, wd1, wbf);
    init_counts<<<1, 32, 0, stream>>>(cntb);
    route_tokens<<<NTOK / 256, 256, 0, stream>>>(mask, idxb, cntb);
    moe_swiglu_wmma<<<dim3(NTOK / M_TILE, 2), 256, LDS_BYTES, stream>>>(
        x, wbf, idxb, cntb, out);
}
